// DecoderRNN_80573586473144
// MI455X (gfx1250) — matrix operand order repaired and checked by reading, not compiled, not run on hardware
//
#include <hip/hip_runtime.h>
#include <hip/hip_bf16.h>
#include <math.h>

// ---------------------------------------------------------------------------
// DecoderRNN greedy decode for MI455X (gfx1250, wave32, WMMA bf16)
// B=64, H=1024, E=512, V=32000, T=31 steps
// ---------------------------------------------------------------------------

#define Bv   64
#define Hv   1024
#define Ev   512
#define Vv   32000
#define Kcat 2560          // E + H + H  (emb | enc_h | h)
#define G4H  4096          // 4*H
#define Tsteps 31
#define NPANEL 32          // W_out rows staged per block in logits GEMM

typedef __attribute__((ext_vector_type(8)))  __bf16 v8bf;
typedef __attribute__((ext_vector_type(16))) __bf16 v16bf;
typedef __attribute__((ext_vector_type(8)))  float  v8f;

__device__ __forceinline__ __bf16 f2bf(float f) {
    union { float f; unsigned u; } v; v.f = f;
    unsigned r = v.u + 0x7FFFu + ((v.u >> 16) & 1u);   // round-to-nearest-even
    unsigned short h = (unsigned short)(r >> 16);
    __bf16 b; __builtin_memcpy(&b, &h, 2); return b;
}

__device__ __forceinline__ float sigmoidf_(float x) {
    return 1.0f / (1.0f + expf(-x));
}

// ---------------------------------------------------------------------------
// Prologue: pack & convert weights to bf16 in workspace
// ---------------------------------------------------------------------------
__global__ __launch_bounds__(256) void k_pack_wcat(const float* __restrict__ Wih,
                                                   const float* __restrict__ Whh,
                                                   __bf16* __restrict__ Wcat) {
    int i = blockIdx.x * 256 + threadIdx.x;            // over 4096*2560
    if (i >= G4H * Kcat) return;
    int n = i / Kcat, k = i - n * Kcat;
    float v = (k < (Ev + Hv)) ? Wih[n * (Ev + Hv) + k]
                              : Whh[n * Hv + (k - (Ev + Hv))];
    Wcat[i] = f2bf(v);
}

__global__ __launch_bounds__(256) void k_f32_to_bf16(const float* __restrict__ src,
                                                     __bf16* __restrict__ dst, int n) {
    int i = blockIdx.x * 256 + threadIdx.x;
    if (i < n) dst[i] = f2bf(src[i]);
}

__global__ __launch_bounds__(256) void k_bias_cat(const float* __restrict__ bih,
                                                  const float* __restrict__ bhh,
                                                  float* __restrict__ bcat) {
    int i = blockIdx.x * 256 + threadIdx.x;
    if (i < G4H) bcat[i] = bih[i] + bhh[i];
}

__global__ __launch_bounds__(256) void k_init_state(const float* __restrict__ enc_h,
                                                    const float* __restrict__ enc_c,
                                                    float* __restrict__ c,
                                                    __bf16* __restrict__ xcat) {
    int i = blockIdx.x * 256 + threadIdx.x;            // over 64*1024
    if (i >= Bv * Hv) return;
    int b = i >> 10, n = i & (Hv - 1);
    c[i] = enc_c[i];
    __bf16 hb = f2bf(enc_h[i]);
    xcat[b * Kcat + Ev + n]      = hb;                 // constant enc_h slice
    xcat[b * Kcat + Ev + Hv + n] = hb;                 // initial hidden h = enc_h
}

__global__ __launch_bounds__(256) void k_init_emb(const float* __restrict__ embedding,
                                                  const int* __restrict__ sos,
                                                  __bf16* __restrict__ xcat) {
    int i = blockIdx.x * 256 + threadIdx.x;            // over 64*512
    if (i >= Bv * Ev) return;
    int b = i >> 9, e = i & (Ev - 1);
    xcat[b * Kcat + e] = f2bf(embedding[(size_t)sos[0] * Ev + e]);
}

// ---------------------------------------------------------------------------
// WMMA fragment helpers (ISA 7.12.2 wave32 layouts)
//   A 16x32 bf16: lane m=lane%16; half=lane/16; V0-3: K = k0+8*half..+7
//                                              V4-7: K = k0+16+8*half..+7
//   B 32x16 bf16 (B = W^T, W row-major): lane n=lane%16; same K order as A
//   C/D 16x16 f32: VGPR g -> (M = g+8*half, N = lane%16)
// ---------------------------------------------------------------------------
__device__ __forceinline__ v16bf load_frag_A(const __bf16* __restrict__ row,
                                             int k0, int half) {
    v8bf lo = *(const v8bf*)(row + k0 + 8 * half);
    v8bf hi = *(const v8bf*)(row + k0 + 16 + 8 * half);
    v16bf a;
#pragma unroll
    for (int i = 0; i < 8; ++i) { a[i] = lo[i]; a[i + 8] = hi[i]; }
    return a;
}

__device__ __forceinline__ v16bf load_frag_B(const __bf16* wrow, int k0, int half) {
    v8bf lo = *(const v8bf*)(wrow + k0 + 8 * half);
    v8bf hi = *(const v8bf*)(wrow + k0 + 16 + 8 * half);
    v16bf b;
#pragma unroll
    for (int i = 0; i < 8; ++i) { b[i] = lo[i]; b[i + 8] = hi[i]; }
    return b;
}

// ---------------------------------------------------------------------------
// Step kernel 1: gates = xcat @ Wcat^T + (b_ih+b_hh)   -> (64, 4096) f32
// grid = 256 blocks (one 16-wide N tile each), block = 128 (4 waves = 4 M tiles)
// Weights are L2-resident bf16 (21 MB); direct global loads pipeline well.
// ---------------------------------------------------------------------------
__global__ __launch_bounds__(128) void k_gates_gemm(const __bf16* __restrict__ xcat,
                                                    const __bf16* __restrict__ Wcat,
                                                    const float* __restrict__ bcat,
                                                    float* __restrict__ gates) {
    const int lane = threadIdx.x & 31;
    const int wave = threadIdx.x >> 5;                 // m-tile 0..3
    const int half = lane >> 4;
    const int mr   = lane & 15;
    const int n0   = blockIdx.x * 16;
    const int m0   = wave * 16;

    const __bf16* arow = xcat + (size_t)(m0 + mr) * Kcat;
    const __bf16* brow = Wcat + (size_t)(n0 + mr) * Kcat;

    const float bias = bcat[n0 + mr];
    v8f acc;
#pragma unroll
    for (int g = 0; g < 8; ++g) acc[g] = bias;

    for (int k0 = 0; k0 < Kcat; k0 += 32) {
        v16bf a = load_frag_A(arow, k0, half);
        v16bf b = load_frag_B(brow, k0, half);
        acc = __builtin_amdgcn_wmma_f32_16x16x32_bf16(
                  false, a, false, b, (short)0, acc, false, false);
    }

    float* out = gates + n0 + mr;
#pragma unroll
    for (int g = 0; g < 8; ++g)
        out[(size_t)(m0 + g + 8 * half) * G4H] = acc[g];
}

// ---------------------------------------------------------------------------
// Step kernel 2: LSTM cell pointwise -> updates c (f32) and h slice of xcat (bf16)
// ---------------------------------------------------------------------------
__global__ __launch_bounds__(256) void k_lstm_cell(const float* __restrict__ gates,
                                                   float* __restrict__ c,
                                                   __bf16* __restrict__ xcat) {
    int i = blockIdx.x * 256 + threadIdx.x;            // over 64*1024
    if (i >= Bv * Hv) return;
    int b = i >> 10, n = i & (Hv - 1);
    const float* g = gates + (size_t)b * G4H;
    float ig = sigmoidf_(g[n]);
    float fg = sigmoidf_(g[Hv + n]);
    float gg = tanhf(g[2 * Hv + n]);
    float og = sigmoidf_(g[3 * Hv + n]);
    float cn = fg * c[i] + ig * gg;
    float hn = og * tanhf(cn);
    c[i] = cn;
    xcat[(size_t)b * Kcat + Ev + Hv + n] = f2bf(hn);
}

// ---------------------------------------------------------------------------
// Step kernel 3: logits = h @ W_out^T + b_out  -> (64, 32000) f32
//
// Block = 256 threads (8 waves) covering a 64x32 output panel:
//   waves = 4 M-tiles x 2 N-tiles over a shared 32-row W_out panel.
// The 64 KB bf16 weight panel (32 rows x K=1024, contiguous in memory) is
// staged into LDS with GLOBAL_LOAD_ASYNC_TO_LDS_B128 (ASYNCcnt), then WMMA
// consumes it via ds_load_b128 — the 4x wave-redundant global B traffic of
// the naive version disappears. grid = 1000 blocks.
// ---------------------------------------------------------------------------
__global__ __launch_bounds__(256) void k_logits_gemm(const __bf16* __restrict__ xcat,
                                                     const __bf16* __restrict__ Wout,
                                                     const float* __restrict__ bout,
                                                     float* __restrict__ logits) {
    __shared__ __align__(16) __bf16 ldsB[NPANEL * Hv];          // 64 KB

    const int tid  = threadIdx.x;
    const int lane = tid & 31;
    const int wave = tid >> 5;                                  // 0..7
    const int half = lane >> 4;
    const int mr   = lane & 15;
    const int n0   = blockIdx.x * NPANEL;

    // ---- async stage: 32 consecutive W_out rows (contiguous 64 KB) -> LDS ----
    {
        const char* gbase = (const char*)(Wout + (size_t)n0 * Hv);
        unsigned    lbase = (unsigned)(uintptr_t)(&ldsB[0]);
#pragma unroll
        for (int j = 0; j < (NPANEL * Hv * 2) / 16 / 256; ++j) {     // 16 chunks/thread
            const int c = tid + j * 256;
            unsigned long long ga = (unsigned long long)(uintptr_t)(gbase + c * 16);
            unsigned           la = lbase + (unsigned)(c * 16);
            asm volatile("global_load_async_to_lds_b128 %0, %1, off"
                         :: "v"(la), "v"(ga) : "memory");
        }
        asm volatile("s_wait_asynccnt 0x0" ::: "memory");
        __syncthreads();
    }

    const int m0 = (wave & 3) * 16;                  // M tile
    const int nl = (wave >> 2) * 16;                 // which staged N tile (0|16)
    const __bf16* arow = xcat + (size_t)(m0 + mr) * Kcat + (Ev + Hv);
    const __bf16* brow = &ldsB[(nl + mr) * Hv];      // LDS -> ds_load_b128

    const float bias = bout[n0 + nl + mr];
    v8f acc;
#pragma unroll
    for (int g = 0; g < 8; ++g) acc[g] = bias;

    for (int k0 = 0; k0 < Hv; k0 += 32) {
        v16bf a = load_frag_A(arow, k0, half);
        v16bf b = load_frag_B(brow, k0, half);
        acc = __builtin_amdgcn_wmma_f32_16x16x32_bf16(
                  false, a, false, b, (short)0, acc, false, false);
    }

    float* out = logits + n0 + nl + mr;
#pragma unroll
    for (int g = 0; g < 8; ++g)
        out[(size_t)(m0 + g + 8 * half) * Vv] = acc[g];
}

// ---------------------------------------------------------------------------
// Step kernel 4: log_softmax + argmax + write outputs + embedding gather
// one block per batch row (64 blocks x 256 threads)
// ---------------------------------------------------------------------------
__global__ __launch_bounds__(256) void k_softmax_argmax(const float* __restrict__ logits,
                                                        const __bf16* __restrict__ emb_bf,
                                                        __bf16* __restrict__ xcat,
                                                        float* __restrict__ d_out,
                                                        int t) {
    __shared__ float sv[256];
    __shared__ int   si[256];

    const int b   = blockIdx.x;
    const int tid = threadIdx.x;
    const float* row = logits + (size_t)b * Vv;

    // pass 1: max + first-argmax
    float vmax = -INFINITY; int amax = 0x7FFFFFFF;
    for (int j = tid; j < Vv; j += 256) {
        float v = row[j];
        if (v > vmax || (v == vmax && j < amax)) { vmax = v; amax = j; }
    }
    sv[tid] = vmax; si[tid] = amax;
    __syncthreads();
    for (int s = 128; s > 0; s >>= 1) {
        if (tid < s) {
            float v2 = sv[tid + s]; int i2 = si[tid + s];
            if (v2 > sv[tid] || (v2 == sv[tid] && i2 < si[tid])) {
                sv[tid] = v2; si[tid] = i2;
            }
        }
        __syncthreads();
    }
    const float m    = sv[0];
    const int   best = si[0];
    __syncthreads();

    // pass 2: sum(exp(x - m))
    float ssum = 0.0f;
    for (int j = tid; j < Vv; j += 256) ssum += expf(row[j] - m);
    sv[tid] = ssum;
    __syncthreads();
    for (int s = 128; s > 0; s >>= 1) {
        if (tid < s) sv[tid] += sv[tid + s];
        __syncthreads();
    }
    const float lse = m + logf(sv[0]);

    // pass 3: write log-probs
    float* outp = d_out + ((size_t)t * Bv + b) * Vv;
    for (int j = tid; j < Vv; j += 256) outp[j] = row[j] - lse;

    // chosen token (stored as float in the flattened tuple output)
    if (tid == 0)
        d_out[(size_t)(Tsteps) * Bv * Vv + (size_t)t * Bv + b] = (float)best;

    // gather next-step embedding (bf16) into xcat[:, 0:E]
    const __bf16* erow = emb_bf + (size_t)best * Ev;
    for (int e = tid; e < Ev; e += 256)
        xcat[(size_t)b * Kcat + e] = erow[e];
}

// ---------------------------------------------------------------------------
// Host launcher
// ---------------------------------------------------------------------------
extern "C" void kernel_launch(void* const* d_in, const int* in_sizes, int n_in,
                              void* d_out, int out_size, void* d_ws, size_t ws_size,
                              hipStream_t stream) {
    const float* enc_h     = (const float*)d_in[0];
    const float* enc_c     = (const float*)d_in[1];
    const float* embedding = (const float*)d_in[2];
    const float* W_ih      = (const float*)d_in[3];
    const float* W_hh      = (const float*)d_in[4];
    const float* b_ih      = (const float*)d_in[5];
    const float* b_hh      = (const float*)d_in[6];
    const float* W_out     = (const float*)d_in[7];
    const float* b_out     = (const float*)d_in[8];
    const int*   sos       = (const int*)d_in[9];
    float* out             = (float*)d_out;

    // workspace carve-up (256B aligned)
    char* ws = (char*)d_ws;
    size_t off = 0;
    auto carve = [&](size_t bytes) -> void* {
        void* p = ws + off;
        off = (off + bytes + 255) & ~(size_t)255;
        return p;
    };
    __bf16* Wcat    = (__bf16*)carve((size_t)G4H * Kcat * 2);   // 21.0 MB
    __bf16* Wout_bf = (__bf16*)carve((size_t)Vv * Hv * 2);      // 65.5 MB
    __bf16* emb_bf  = (__bf16*)carve((size_t)Vv * Ev * 2);      // 32.8 MB
    __bf16* xcat    = (__bf16*)carve((size_t)Bv * Kcat * 2);
    float*  cbuf    = (float*)carve((size_t)Bv * Hv * 4);
    float*  gates   = (float*)carve((size_t)Bv * G4H * 4);
    float*  logits  = (float*)carve((size_t)Bv * Vv * 4);       // 8.2 MB
    float*  bcat    = (float*)carve((size_t)G4H * 4);

    // ---- prologue: bf16 conversions + state init ----
    {
        int n;
        n = G4H * Kcat;
        k_pack_wcat<<<(n + 255) / 256, 256, 0, stream>>>(W_ih, W_hh, Wcat);
        n = Vv * Hv;
        k_f32_to_bf16<<<(n + 255) / 256, 256, 0, stream>>>(W_out, Wout_bf, n);
        n = Vv * Ev;
        k_f32_to_bf16<<<(n + 255) / 256, 256, 0, stream>>>(embedding, emb_bf, n);
        k_bias_cat<<<(G4H + 255) / 256, 256, 0, stream>>>(b_ih, b_hh, bcat);
        n = Bv * Hv;
        k_init_state<<<(n + 255) / 256, 256, 0, stream>>>(enc_h, enc_c, cbuf, xcat);
        n = Bv * Ev;
        k_init_emb<<<(n + 255) / 256, 256, 0, stream>>>(embedding, sos, xcat);
    }

    // ---- 31 sequential decode steps ----
    for (int t = 0; t < Tsteps; ++t) {
        k_gates_gemm<<<G4H / 16, 128, 0, stream>>>(xcat, Wcat, bcat, gates);
        k_lstm_cell<<<(Bv * Hv + 255) / 256, 256, 0, stream>>>(gates, cbuf, xcat);
        k_logits_gemm<<<Vv / NPANEL, 256, 0, stream>>>(xcat, Wout_bf, b_out, logits);
        k_softmax_argmax<<<Bv, 256, 0, stream>>>(logits, emb_bf, xcat, out, t);
    }

    (void)in_sizes; (void)n_in; (void)out_size; (void)ws_size;
}
